// SparseConvNet_11441792876794
// MI455X (gfx1250) — matrix operand order repaired and checked by reading, not compiled, not run on hardware
//
#include <hip/hip_runtime.h>
#include <math.h>

// ---------------------------------------------------------------------------
// MI455X (gfx1250) implementation of the sparse-conv feature extractor.
//
// Conv layers are implicit GEMMs on V_WMMA_F32_16X16X32_F16 (wave32):
//   M = 32 output voxels (2 D-tiles sharing one B fragment),
//   N = 16 output channels, K = 32 input channels per step.
// Activations: channels-last f16  [voxel][Cpad]  (Cpad = max(C,32)).
// Weights:     f16 [tap][Cout][CinPad].
//
// Per-lane WMMA fragment layouts (CDNA5 ISA 7.12.2, 16-bit):
//   A 16x32:  lane = m + 16*half holds A[m][half*16 .. +15]  (contiguous k; not the ISA table order)
//   B 32x16:  lane holds B[half*16 .. +15][n = lane&15]  (same k order as A, not ISA table order; contiguous in our layout)
//   C/D f32:  acc[v] is D[m = v + 8*half][n = lane&15]
// ---------------------------------------------------------------------------

typedef __attribute__((ext_vector_type(16))) _Float16 v16h;
typedef __attribute__((ext_vector_type(8)))  _Float16 v8h;
typedef __attribute__((ext_vector_type(8)))  float    v8f;

#define WPB 8  // waves per block (256 threads)

// ---------------- input conversion: NCDHW f32 -> [voxel][Cpad] f16 ----------
__global__ void convert_x_k(const float* __restrict__ x, _Float16* __restrict__ out,
                            int Nvox, int C, int Cpad) {
    int tid = blockIdx.x * blockDim.x + threadIdx.x;
    if (tid >= Nvox * Cpad) return;
    int c   = tid % Cpad;
    int vox = tid / Cpad;
    float v = (c < C) ? x[(size_t)c * Nvox + vox] : 0.0f;
    out[tid] = (_Float16)v;
}

// ---------------- weight conversion: OIDHW f32 -> [tap][O][Ipad] f16 --------
__global__ void convert_w_k(const float* __restrict__ w, _Float16* __restrict__ out,
                            int O, int I, int taps, int Ipad) {
    int tid = blockIdx.x * blockDim.x + threadIdx.x;
    int total = taps * O * Ipad;
    if (tid >= total) return;
    int i = tid % Ipad;
    int r = tid / Ipad;
    int o = r % O;
    int t = r / O;
    float v = (i < I) ? w[((size_t)o * I + i) * taps + t] : 0.0f;
    out[tid] = (_Float16)v;
}

// ---------------- mask 2x2x2 max pool ---------------------------------------
__global__ void pool_mask_k(const float* __restrict__ in, float* __restrict__ out, int Do) {
    int tid = blockIdx.x * blockDim.x + threadIdx.x;
    int N = Do * Do * Do;
    if (tid >= N) return;
    int d = tid / (Do * Do);
    int r = tid - d * Do * Do;
    int h = r / Do;
    int w = r - h * Do;
    int Di = Do * 2;
    float m = 0.0f;
#pragma unroll
    for (int a = 0; a < 2; ++a)
#pragma unroll
        for (int b = 0; b < 2; ++b)
#pragma unroll
            for (int c = 0; c < 2; ++c)
                m = fmaxf(m, in[((size_t)(2 * d + a) * Di + (2 * h + b)) * Di + (2 * w + c)]);
    out[tid] = m;
}

// ---------------- implicit-GEMM WMMA conv (fully shape-templated) -----------
// out = relu( ((conv(in)*mask) * scale + bias) * mask ), mask/BN compile-time.
template <int DIN, int DOUT, int CINP, int COUT, int COUTP,
          int KSZ, int STRIDE, int PAD, bool MASKED, bool BN>
__global__ void __launch_bounds__(256)
conv_wmma_t(const _Float16* __restrict__ in, const _Float16* __restrict__ wts,
            _Float16* __restrict__ out,
            const float* __restrict__ mask,
            const float* __restrict__ gamma, const float* __restrict__ beta) {
    constexpr int NVOX = DOUT * DOUT * DOUT;
    constexpr int TAPS = KSZ * KSZ * KSZ;
    const int lane = threadIdx.x & 31;
    const int wave = threadIdx.x >> 5;
    const int half = lane >> 4;       // 0 or 1
    const int mr   = lane & 15;

    const int vbase = (blockIdx.x * WPB + wave) * 32;   // 2 x 16-voxel tiles
    if (vbase >= NVOX) return;                           // wave-uniform

    const int n = blockIdx.y * 16 + mr;                  // output channel

    // Padded out-channel tiles (stage 0 only): store zeros. Uniform branch.
    if constexpr (COUT < COUTP) {
        if (blockIdx.y * 16 >= COUT) {
#pragma unroll
            for (int tt = 0; tt < 2; ++tt)
#pragma unroll
                for (int v = 0; v < 8; ++v) {
                    int ov = vbase + tt * 16 + half * 8 + v;
                    if (ov < NVOX) out[(size_t)ov * COUTP + n] = (_Float16)0.0f;
                }
            return;
        }
    }

    // Decode this lane's two A-row voxels (compile-time DOUT -> mul/shift).
    bool vv[2];
    int d[2], h[2], w[2];
#pragma unroll
    for (int tt = 0; tt < 2; ++tt) {
        int vox = vbase + tt * 16 + mr;
        vv[tt] = vox < NVOX;
        int t = vv[tt] ? vox : 0;
        d[tt] = t / (DOUT * DOUT);
        int r = t - d[tt] * DOUT * DOUT;
        h[tt] = r / DOUT;
        w[tt] = r - h[tt] * DOUT;
    }

    v8f accs[2] = {{}, {}};
    const v8h zero8 = {};

#pragma unroll 1
    for (int tap = 0; tap < TAPS; ++tap) {
        const int kd = tap / (KSZ * KSZ);
        const int rr = tap - kd * KSZ * KSZ;
        const int kh = rr / KSZ;
        const int kw = rr - kh * KSZ;

        bool ok[2];
        size_t srow[2];
#pragma unroll
        for (int tt = 0; tt < 2; ++tt) {
            int sd = d[tt] * STRIDE + kd - PAD;
            int sh = h[tt] * STRIDE + kh - PAD;
            int sw = w[tt] * STRIDE + kw - PAD;
            ok[tt] = vv[tt] & (sd >= 0) & (sd < DIN) & (sh >= 0) & (sh < DIN) &
                     (sw >= 0) & (sw < DIN);
            srow[tt] = ((size_t)(sd * DIN + sh) * DIN + sw) * (size_t)CINP;
        }
        const _Float16* wrow = wts + ((size_t)tap * COUT + n) * CINP + half * 16;

#pragma unroll
        for (int k0 = 0; k0 < CINP; k0 += 32) {
            // B fragment loaded ONCE, feeds two WMMAs (M=32 per wave).
            v16h b = *(const v16h*)(wrow + k0);
#pragma unroll
            for (int tt = 0; tt < 2; ++tt) {
                union { v16h v; v8h h8[2]; } a;
                if (ok[tt]) {
                    a.h8[0] = *(const v8h*)(in + srow[tt] + k0 + half * 16);
                    a.h8[1] = *(const v8h*)(in + srow[tt] + k0 + half * 16 + 8);
                } else {
                    a.h8[0] = zero8;
                    a.h8[1] = zero8;
                }
                // v_wmma_f32_16x16x32_f16
                accs[tt] = __builtin_amdgcn_wmma_f32_16x16x32_f16(
                    false, a.v, false, b, (short)0, accs[tt], false, false);
            }
        }
    }

    // Fused epilogue: mask, BN (eval: scale = gamma*rsqrt(1+eps)), mask, ReLU.
    float sc = 1.0f, bs = 0.0f;
    if constexpr (BN) { sc = gamma[n] * rsqrtf(1.0f + 1e-3f); bs = beta[n]; }
#pragma unroll
    for (int tt = 0; tt < 2; ++tt)
#pragma unroll
        for (int v = 0; v < 8; ++v) {
            int ov = vbase + tt * 16 + half * 8 + v;   // D row m = v + 8*half
            if (ov >= NVOX) continue;
            float mk = 1.0f;
            if constexpr (MASKED) mk = mask[ov];
            float val = accs[tt][v] * mk;
            val = val * sc + bs;
            if constexpr (MASKED) val *= mk;
            val = fmaxf(val, 0.0f);
            out[(size_t)ov * COUTP + n] = (_Float16)val;
        }
}

// ---------------- trilinear grid sample (align_corners=True, zeros pad) -----
__global__ void gridsample_k(const _Float16* __restrict__ vol, int D, int Cpad, int C,
                             const float* __restrict__ grid, int npts,
                             _Float16* __restrict__ outH,
                             float* __restrict__ outF, int outStride) {
    int nG = C >> 3;
    int tid = blockIdx.x * blockDim.x + threadIdx.x;
    if (tid >= npts * nG) return;
    int cg = tid % nG;
    int p  = tid / nG;

    float gx = grid[(size_t)p * 3 + 0];
    float gy = grid[(size_t)p * 3 + 1];
    float gz = grid[(size_t)p * 3 + 2];
    float fx = (gx + 1.0f) * 0.5f * (float)(D - 1);
    float fy = (gy + 1.0f) * 0.5f * (float)(D - 1);
    float fz = (gz + 1.0f) * 0.5f * (float)(D - 1);
    float x0 = floorf(fx), y0 = floorf(fy), z0 = floorf(fz);
    float tx = fx - x0, ty = fy - y0, tz = fz - z0;

    float acc[8] = {0, 0, 0, 0, 0, 0, 0, 0};
#pragma unroll
    for (int dz = 0; dz < 2; ++dz)
#pragma unroll
        for (int dy = 0; dy < 2; ++dy)
#pragma unroll
            for (int dx = 0; dx < 2; ++dx) {
                float xi = x0 + dx, yi = y0 + dy, zi = z0 + dz;
                bool valid = (xi >= 0.0f) & (xi < (float)D) & (yi >= 0.0f) &
                             (yi < (float)D) & (zi >= 0.0f) & (zi < (float)D);
                float wgt = (dx ? tx : 1.0f - tx) * (dy ? ty : 1.0f - ty) *
                            (dz ? tz : 1.0f - tz);
                if (valid) {
                    int xc = (int)xi, yc = (int)yi, zc = (int)zi;
                    size_t base = ((size_t)(zc * D + yc) * D + xc) * Cpad + cg * 8;
                    v8h vals = *(const v8h*)(vol + base);
#pragma unroll
                    for (int i = 0; i < 8; ++i) acc[i] += wgt * (float)vals[i];
                }
            }

    if (outH) {
#pragma unroll
        for (int i = 0; i < 8; ++i)
            outH[(size_t)p * Cpad + cg * 8 + i] = (_Float16)acc[i];
    } else {
#pragma unroll
        for (int i = 0; i < 8; ++i)
            outF[(size_t)(cg * 8 + i) * outStride + p] = acc[i];
    }
}

// ---------------------------------------------------------------------------
// Host orchestration
// ---------------------------------------------------------------------------
static inline size_t alignUp(size_t v) { return (v + 255) & ~(size_t)255; }

template <int DIN, int DOUT, int CINP, int COUT, int COUTP,
          int KSZ, int STRIDE, int PAD, bool MASKED, bool BN>
static void run_conv(hipStream_t s, const _Float16* in, const _Float16* w,
                     _Float16* out, const float* mask, const float* g,
                     const float* b) {
    constexpr int NVOX = DOUT * DOUT * DOUT;
    int tiles = (NVOX + 31) / 32;
    dim3 grid((tiles + WPB - 1) / WPB, COUTP / 16);
    conv_wmma_t<DIN, DOUT, CINP, COUT, COUTP, KSZ, STRIDE, PAD, MASKED, BN>
        <<<grid, 256, 0, s>>>(in, w, out, mask, g, b);
}

extern "C" void kernel_launch(void* const* d_in, const int* in_sizes, int n_in,
                              void* d_out, int out_size, void* d_ws, size_t ws_size,
                              hipStream_t stream) {
    (void)out_size; (void)ws_size;

    // ---- bind inputs by flat element count (robust to dict vs pytree order).
    const float* W[17] = {};  // w0a w0b wd0 w1a w1b wd1 w2a w2b w2c wd2 w3a w3b w3c wd3 w4a w4b w4c
    const float* G[9]  = {};
    const float* Bt[9] = {};
    const float *xin = nullptr, *mask0 = nullptr, *tg = nullptr, *pg = nullptr;
    int c6912 = 0, c27648 = 0, c110592 = 0, c442368 = 0, c64 = 0, c128 = 0;
    for (int i = 0; i < n_in; ++i) {
        const float* p = (const float*)d_in[i];
        switch (in_sizes[i]) {
            case 6912:    W[c6912 < 1 ? 0 : 1] = p; c6912++; break;
            case 4096:    W[2] = p; break;
            case 27648:   W[3 + (c27648 < 2 ? c27648 : 1)] = p; c27648++; break;
            case 16384:   W[5] = p; break;
            case 110592:  if (c110592 < 3) W[6 + c110592] = p; c110592++; break;
            case 65536:   W[9] = p; break;
            case 442368:  if (c442368 < 6) W[c442368 < 3 ? 10 + c442368 : 11 + c442368] = p;
                          c442368++; break;
            case 131072:  W[13] = p; break;
            case 64:      if (c64 < 6) { if ((c64 & 1) == 0) G[c64 >> 1] = p; else Bt[c64 >> 1] = p; }
                          c64++; break;
            case 128:     if (c128 < 12) { int j = c128 >> 1;
                              if ((c128 & 1) == 0) G[3 + j] = p; else Bt[3 + j] = p; }
                          c128++; break;
            case 14155776: xin = p; break;
            case 884736:   mask0 = p; break;
            case 98304:    tg = p; break;
            case 196608:   pg = p; break;
            default: break;
        }
    }
    if (!xin || !mask0 || !tg || !pg || !W[16]) return;  // binding failed

    float* outF = (float*)d_out;

    // ---- workspace layout (~123 MB). A0/B0 are 96^3 x 32ch f16 ping-pong;
    // all lower-resolution buffers overlay B0 (B0 dead after stage 0).
    char* ws = (char*)d_ws;
    const size_t SZ_A0 = (size_t)884736 * 32 * 2;  // 56,623,104 B
    _Float16* A0 = (_Float16*)ws;
    char* ovl = ws + SZ_A0;
    _Float16* B0 = (_Float16*)ovl;
    size_t oo = 0;
    auto sub = [&](size_t bytes) { _Float16* q = (_Float16*)(ovl + oo); oo += alignUp(bytes); return q; };
    _Float16* A1 = sub((size_t)110592 * 32 * 2);
    _Float16* B1 = sub((size_t)110592 * 32 * 2);
    _Float16* A2 = sub((size_t)13824 * 64 * 2);
    _Float16* B2 = sub((size_t)13824 * 64 * 2);
    _Float16* A3 = sub((size_t)1728 * 128 * 2);
    _Float16* B3 = sub((size_t)1728 * 128 * 2);
    _Float16* A4 = sub((size_t)216 * 128 * 2);
    _Float16* B4 = sub((size_t)216 * 128 * 2);

    // f16 weights, mask pyramid, grid-sample temp after the two big buffers.
    char* wreg = ws + 2 * SZ_A0;
    struct WInfo { int O, I, taps, Ipad; };
    const WInfo wi[17] = {
        {16, 16, 27, 32},   {16, 16, 27, 32},   {32, 16, 8, 32},
        {32, 32, 27, 32},   {32, 32, 27, 32},   {64, 32, 8, 32},
        {64, 64, 27, 64},   {64, 64, 27, 64},   {64, 64, 27, 64},   {128, 64, 8, 64},
        {128, 128, 27, 128},{128, 128, 27, 128},{128, 128, 27, 128},{128, 128, 8, 128},
        {128, 128, 27, 128},{128, 128, 27, 128},{128, 128, 27, 128}};
    _Float16* WH[17];
    size_t wo = 0;
    for (int i = 0; i < 17; ++i) {
        WH[i] = (_Float16*)(wreg + wo);
        wo += alignUp((size_t)wi[i].taps * wi[i].O * wi[i].Ipad * 2);
    }
    char* mreg = wreg + wo;
    float* m1 = (float*)mreg;                          // 48^3
    float* m2 = (float*)(mreg + alignUp(110592 * 4));  // 24^3
    float* m3 = (float*)((char*)m2 + alignUp(13824 * 4));
    float* m4 = (float*)((char*)m3 + alignUp(1728 * 4));
    _Float16* T = (_Float16*)((char*)m4 + alignUp(216 * 4));  // [32768][<=128] f16

    // ---- weight conversion + input conversion + mask pyramid.
    for (int i = 0; i < 17; ++i) {
        int tot = wi[i].taps * wi[i].O * wi[i].Ipad;
        convert_w_k<<<(tot + 255) / 256, 256, 0, stream>>>(W[i], WH[i], wi[i].O,
                                                           wi[i].I, wi[i].taps,
                                                           wi[i].Ipad);
    }
    {
        int tot = 884736 * 32;
        convert_x_k<<<(tot + 255) / 256, 256, 0, stream>>>(xin, A0, 884736, 16, 32);
    }
    pool_mask_k<<<(110592 + 255) / 256, 256, 0, stream>>>(mask0, m1, 48);
    pool_mask_k<<<(13824 + 255) / 256, 256, 0, stream>>>(m1, m2, 24);
    pool_mask_k<<<(1728 + 255) / 256, 256, 0, stream>>>(m2, m3, 12);
    pool_mask_k<<<(216 + 255) / 256, 256, 0, stream>>>(m3, m4, 6);

    auto sample_stage = [&](const _Float16* net, int D, int C, int rowOff) {
        int nG = C / 8;
        int t1 = 32768 * nG;
        gridsample_k<<<(t1 + 255) / 256, 256, 0, stream>>>(net, D, C, C, tg, 32768,
                                                           T, nullptr, 0);
        int t2 = 65536 * nG;
        gridsample_k<<<(t2 + 255) / 256, 256, 0, stream>>>(T, 32, C, C, pg, 65536,
                                                           nullptr,
                                                           outF + (size_t)rowOff * 65536,
                                                           65536);
    };

    // ---- stage 0 @96^3, C=16 (pad 32)
    run_conv<96, 96, 32, 16, 32, 3, 1, 1, true, false>(stream, A0, WH[0], B0, mask0, nullptr, nullptr);
    run_conv<96, 96, 32, 16, 32, 3, 1, 1, true, false>(stream, B0, WH[1], A0, mask0, nullptr, nullptr);
    run_conv<96, 48, 32, 32, 32, 2, 2, 0, false, false>(stream, A0, WH[2], A1, nullptr, nullptr, nullptr);

    // ---- stage 1 @48^3, C=32
    run_conv<48, 48, 32, 32, 32, 3, 1, 1, true, false>(stream, A1, WH[3], B1, m1, nullptr, nullptr);
    run_conv<48, 48, 32, 32, 32, 3, 1, 1, true, false>(stream, B1, WH[4], A1, m1, nullptr, nullptr);
    sample_stage(A1, 48, 32, 0);                                   // f1 rows 0..31
    run_conv<48, 24, 32, 64, 64, 2, 2, 0, false, false>(stream, A1, WH[5], A2, nullptr, nullptr, nullptr);

    // ---- stage 2 @24^3, C=64, BN
    run_conv<24, 24, 64, 64, 64, 3, 1, 1, true, true>(stream, A2, WH[6], B2, m2, G[0], Bt[0]);
    run_conv<24, 24, 64, 64, 64, 3, 1, 1, true, true>(stream, B2, WH[7], A2, m2, G[1], Bt[1]);
    run_conv<24, 24, 64, 64, 64, 3, 1, 1, true, true>(stream, A2, WH[8], B2, m2, G[2], Bt[2]);
    sample_stage(B2, 24, 64, 32);                                  // f2 rows 32..95
    run_conv<24, 12, 64, 128, 128, 2, 2, 0, false, false>(stream, B2, WH[9], A3, nullptr, nullptr, nullptr);

    // ---- stage 3 @12^3, C=128, BN
    run_conv<12, 12, 128, 128, 128, 3, 1, 1, true, true>(stream, A3, WH[10], B3, m3, G[3], Bt[3]);
    run_conv<12, 12, 128, 128, 128, 3, 1, 1, true, true>(stream, B3, WH[11], A3, m3, G[4], Bt[4]);
    run_conv<12, 12, 128, 128, 128, 3, 1, 1, true, true>(stream, A3, WH[12], B3, m3, G[5], Bt[5]);
    sample_stage(B3, 12, 128, 96);                                 // f3 rows 96..223
    run_conv<12, 6, 128, 128, 128, 2, 2, 0, false, false>(stream, B3, WH[13], A4, nullptr, nullptr, nullptr);

    // ---- stage 4 @6^3, C=128, BN
    run_conv<6, 6, 128, 128, 128, 3, 1, 1, true, true>(stream, A4, WH[14], B4, m4, G[6], Bt[6]);
    run_conv<6, 6, 128, 128, 128, 3, 1, 1, true, true>(stream, B4, WH[15], A4, m4, G[7], Bt[7]);
    run_conv<6, 6, 128, 128, 128, 3, 1, 1, true, true>(stream, A4, WH[16], B4, m4, G[8], Bt[8]);
    sample_stage(B4, 6, 128, 224);                                 // f4 rows 224..351
}
